// GRIDGRU_7868380086673
// MI455X (gfx1250) — matrix operand order repaired and checked by reading, not compiled, not run on hardware
//
#include <hip/hip_runtime.h>
#include <stdint.h>

#define DEV __device__ __forceinline__

typedef __attribute__((ext_vector_type(16))) __bf16 v16bf;
typedef __attribute__((ext_vector_type(8)))  float  v8f;

namespace {
constexpr int NB = 32;           // batch
constexpr int TT = 512;          // time steps
constexpr int DD = 1024;         // input dim
constexpr int HH = 1024;         // hidden dim
constexpr int DH = DD + HH;      // 2048 = weight rows
constexpr int WC = 3 * DH;       // 6144 = weight cols
constexpr int G3 = 3 * HH;       // 3072 = temporal gates width
constexpr int NT = NB * TT;      // 16384 = big-GEMM M
}

// ---------------- WMMA fragment helpers (bf16 16x16x32, f32 acc) ----------
// A (16x32, 16-bit): lane l (m=l&15, h=l>>4): VGPR0-7 hold K=16h..16h+15.
DEV v16bf ld_a(const __bf16* rowk, int half) {
  union { uint4 q[2]; v16bf v; } u;
  u.q[0] = *(const uint4*)(rowk + 16 * half);
  u.q[1] = *(const uint4*)(rowk + 16 * half + 8);
  return u.v;
}
// B (32x16, 16-bit): lane l (n=l&15, h=l>>4): VGPR0-7 hold K=16h..16h+15.
DEV v16bf ld_b(const __bf16* colk, int half) {
  union { uint4 q[2]; v16bf v; } u;
  const uint4* p = (const uint4*)(colk + 16 * half);
  u.q[0] = p[0];
  u.q[1] = p[1];
  return u.v;
}
DEV v8f wmma_bf16(v16bf a, v16bf b, v8f c) {
  return __builtin_amdgcn_wmma_f32_16x16x32_bf16(false, a, false, b, (short)0, c,
                                                 false, false);
}
DEV float sigmf(float z) { return 1.0f / (1.0f + __expf(-z)); }

// ---------------- CDNA5 async global->LDS copy (ASYNCcnt) -----------------
// GLOBAL_LOAD_ASYNC_TO_LDS_B128: VDST = VGPR holding LDS byte address,
// VADDR = 64-bit global address, SADDR = off (GV mode).
DEV void async_g2l_b128(uint32_t lds_addr, const void* gaddr) {
  asm volatile("global_load_async_to_lds_b128 %0, %1, off"
               :: "v"(lds_addr), "v"((unsigned long long)(uintptr_t)gaddr)
               : "memory");
}
DEV void wait_async_barrier() {
  asm volatile("s_wait_asynccnt 0x0" ::: "memory");
  __syncthreads();
}
DEV uint32_t lds_addr32(const void* p) { return (uint32_t)(uintptr_t)p; }

// ---------------- LDS-staged block GEMM core ------------------------------
// Block: 256 threads = 8 waves as 2(M) x 4(N). Block tile 64 x 256, k-step 32.
// Wave tile: 32 x 64 = 8 accumulators (8 WMMAs per k-step).
// LDS: A 64x32 bf16 (4KB), B 256x32 bf16 (16KB), double-buffered (40KB).
DEV void issue_tile(const __bf16* Abase, size_t Astride, const __bf16* Bbase,
                    int k0, char* lA, char* lB, int tid) {
  {  // A tile: 64 rows x 64B/row = 256 chunks of 16B
    int row = tid >> 2, ch = tid & 3;
    const __bf16* g = Abase + (size_t)row * Astride + k0 + ch * 8;
    async_g2l_b128(lds_addr32(lA + row * 64 + ch * 16), g);
  }
#pragma unroll
  for (int it = 0; it < 4; ++it) {  // B tile: 256 cols x 64B/col = 1024 chunks
    int idx = it * 256 + tid;
    int col = idx >> 2, ch = idx & 3;
    const __bf16* g = Bbase + (size_t)col * DH + k0 + ch * 8;
    async_g2l_b128(lds_addr32(lB + col * 64 + ch * 16), g);
  }
}

DEV void compute_tile(v8f (&acc)[2][4], const char* lA, const char* lB,
                      int wm, int wn, int l, int h) {
  v16bf af[2], bf[4];
#pragma unroll
  for (int i = 0; i < 2; ++i) {  // A frags: rows wm*32 + i*16 + l
    const char* p = lA + (wm * 32 + i * 16 + l) * 64;
    union { uint4 q[2]; v16bf v; } u;
    u.q[0] = *(const uint4*)(p + 32 * h);
    u.q[1] = *(const uint4*)(p + 32 * h + 16);
    af[i] = u.v;
  }
#pragma unroll
  for (int j = 0; j < 4; ++j) {  // B frags: cols wn*64 + j*16 + l
    const char* p = lB + (wn * 64 + j * 16 + l) * 64 + 32 * h;
    union { uint4 q[2]; v16bf v; } u;
    u.q[0] = *(const uint4*)(p);
    u.q[1] = *(const uint4*)(p + 16);
    bf[j] = u.v;
  }
#pragma unroll
  for (int i = 0; i < 2; ++i)
#pragma unroll
    for (int j = 0; j < 4; ++j)
      acc[i][j] = wmma_bf16(af[i], bf[j], acc[i][j]);
}

// One K-segment, double-buffered: async loads of k+32 overlap WMMA on k.
DEV void gemm_seg(v8f (&acc)[2][4], const __bf16* Abase, size_t Astride,
                  const __bf16* Bbase, int K, int tid, int wm, int wn, int l,
                  int h, char (*sA)[4096], char (*sB)[16384]) {
  issue_tile(Abase, Astride, Bbase, 0, sA[0], sB[0], tid);
  wait_async_barrier();
  int buf = 0;
  for (int k = 32; k < K; k += 32) {
    issue_tile(Abase, Astride, Bbase, k, sA[buf ^ 1], sB[buf ^ 1], tid);
    compute_tile(acc, sA[buf], sB[buf], wm, wn, l, h);
    wait_async_barrier();
    buf ^= 1;
  }
  compute_tile(acc, sA[buf], sB[buf], wm, wn, l, h);
  __syncthreads();  // buffers reusable by a following segment
}

// ---------------- prep kernels --------------------------------------------
__global__ __launch_bounds__(256) void k_wt(const float* __restrict__ w,
                                            __bf16* __restrict__ wt) {
  size_t i = (size_t)blockIdx.x * 256 + threadIdx.x;
  if (i >= (size_t)WC * DH) return;
  size_t col = i / DH, row = i % DH;
  wt[i] = (__bf16)w[row * (size_t)WC + col];
}

__global__ __launch_bounds__(256) void k_f2b(const float* __restrict__ in,
                                             __bf16* __restrict__ out, size_t n) {
  size_t i = (size_t)blockIdx.x * 256 + threadIdx.x;
  if (i < n) out[i] = (__bf16)in[i];
}

// ---------------- phase 1: gates = x @ Wxt + b_t (LDS-staged) --------------
__global__ __launch_bounds__(256) void k_gates(const __bf16* __restrict__ xb,
                                               const __bf16* __restrict__ wt,
                                               const float* __restrict__ bias,
                                               float* __restrict__ gates) {
  __shared__ __align__(16) char sA[2][4096];
  __shared__ __align__(16) char sB[2][16384];
  int tid = threadIdx.x, wave = tid >> 5, lane = tid & 31;
  int wm = wave & 1, wn = wave >> 1, l = lane & 15, h = lane >> 4;
  int row0 = blockIdx.y * 64, col0 = blockIdx.x * 256;
  v8f acc[2][4] = {};
  gemm_seg(acc, xb + (size_t)row0 * DD, DD, wt + (size_t)col0 * DH, DD, tid, wm,
           wn, l, h, sA, sB);
#pragma unroll
  for (int i = 0; i < 2; ++i)
#pragma unroll
    for (int j = 0; j < 4; ++j) {
      int cc = col0 + wn * 64 + j * 16 + l;
      float bv = bias[cc];
#pragma unroll
      for (int v = 0; v < 8; ++v) {
        int rr = row0 + wm * 32 + i * 16 + v + 8 * h;
        gates[(size_t)rr * G3 + cc] = acc[i][j][v] + bv;
      }
    }
}

// ---------------- phase 2: recurrent scan (direct-register WMMA) ----------
__global__ __launch_bounds__(256) void k_scang(int t,
                                               const __bf16* __restrict__ hb,
                                               size_t hs,
                                               const float* __restrict__ hf,
                                               const __bf16* __restrict__ wt,
                                               const float* __restrict__ gates,
                                               float* __restrict__ u,
                                               __bf16* __restrict__ rh) {
  int wave = threadIdx.x >> 5, lane = threadIdx.x & 31;
  int half = lane >> 4, l = lane & 15;
  int row0 = (wave & 1) * 16;
  int col0 = blockIdx.x * 64 + (wave >> 1) * 16;
  const __bf16* Arow = hb + (size_t)(row0 + l) * hs;
  const __bf16* Bcol = wt + (size_t)(col0 + l) * DH + DD;  // Whtg rows D..D+H-1
  v8f acc = {};
  for (int k = 0; k < HH; k += 32) {
    __builtin_prefetch(Bcol + k + 256, 0, 3);
    acc = wmma_bf16(ld_a(Arow + k, half), ld_b(Bcol + k, half), acc);
  }
  int c = col0 + l;
#pragma unroll
  for (int v = 0; v < 8; ++v) {
    int n = row0 + v + 8 * half;  // batch index
    float g = sigmf(acc[v] + gates[((size_t)n * TT + t) * G3 + c]);
    if (c < HH) {
      u[n * HH + c] = g;
    } else {
      float prev = hf[(size_t)n * hs + (c - HH)];
      rh[n * HH + (c - HH)] = (__bf16)(g * prev);
    }
  }
}

__global__ __launch_bounds__(256) void k_scanh(int t,
                                               const __bf16* __restrict__ rh,
                                               const float* __restrict__ u,
                                               const float* __restrict__ hf,
                                               size_t hs,
                                               const __bf16* __restrict__ wt,
                                               const float* __restrict__ gates,
                                               float* __restrict__ htf,
                                               __bf16* __restrict__ htb) {
  int wave = threadIdx.x >> 5, lane = threadIdx.x & 31;
  int half = lane >> 4, l = lane & 15;
  int row0 = (wave & 1) * 16;
  int col0 = blockIdx.x * 64 + (wave >> 1) * 16;
  const __bf16* Arow = rh + (size_t)(row0 + l) * HH;
  const __bf16* Bcol = wt + (size_t)(2 * HH + col0 + l) * DH + DD;  // Whtc
  v8f acc = {};
  for (int k = 0; k < HH; k += 32) {
    __builtin_prefetch(Bcol + k + 256, 0, 3);
    acc = wmma_bf16(ld_a(Arow + k, half), ld_b(Bcol + k, half), acc);
  }
  int c = col0 + l;
#pragma unroll
  for (int v = 0; v < 8; ++v) {
    int n = row0 + v + 8 * half;
    float hc = tanhf(acc[v] + gates[((size_t)n * TT + t) * G3 + (2 * HH + c)]);
    float prev = hf[(size_t)n * hs + c];
    float uu = u[n * HH + c];
    float nxt = prev + uu * (hc - prev);
    size_t o = ((size_t)n * TT + t) * HH + c;  // row (n*T+t) layout
    htf[o] = nxt;
    htb[o] = (__bf16)nxt;
  }
}

// ---------------- phase 3: depth gates (LDS-staged, 2 K-segments) ---------
__global__ __launch_bounds__(256) void k_post1(const __bf16* __restrict__ xb,
                                               const __bf16* __restrict__ htb,
                                               const float* __restrict__ x,
                                               const __bf16* __restrict__ wt,
                                               const float* __restrict__ bias,
                                               float* __restrict__ ud,
                                               __bf16* __restrict__ rx) {
  __shared__ __align__(16) char sA[2][4096];
  __shared__ __align__(16) char sB[2][16384];
  int tid = threadIdx.x, wave = tid >> 5, lane = tid & 31;
  int wm = wave & 1, wn = wave >> 1, l = lane & 15, h = lane >> 4;
  int row0 = blockIdx.y * 64, col0 = blockIdx.x * 256;
  const __bf16* Bbase = wt + (size_t)(3 * HH + col0) * DH;
  v8f acc[2][4] = {};
  gemm_seg(acc, xb + (size_t)row0 * DD, DD, Bbase, DD, tid, wm, wn, l, h, sA, sB);
  gemm_seg(acc, htb + (size_t)row0 * HH, HH, Bbase + DD, HH, tid, wm, wn, l, h,
           sA, sB);
#pragma unroll
  for (int i = 0; i < 2; ++i)
#pragma unroll
    for (int j = 0; j < 4; ++j) {
      int c = col0 + wn * 64 + j * 16 + l;
      float bv = bias[3 * HH + c];
#pragma unroll
      for (int v = 0; v < 8; ++v) {
        int r = row0 + wm * 32 + i * 16 + v + 8 * h;
        float g = sigmf(acc[i][j][v] + bv);
        if (c < DD) {
          ud[(size_t)r * DD + c] = g;
        } else {
          float xv = x[(size_t)r * DD + (c - DD)];
          rx[(size_t)r * DD + (c - DD)] = (__bf16)(g * xv);
        }
      }
    }
}

__global__ __launch_bounds__(256) void k_post2(const __bf16* __restrict__ htb,
                                               const __bf16* __restrict__ rx,
                                               const float* __restrict__ x,
                                               const float* __restrict__ ud,
                                               const __bf16* __restrict__ wt,
                                               const float* __restrict__ bias,
                                               float* __restrict__ out) {
  __shared__ __align__(16) char sA[2][4096];
  __shared__ __align__(16) char sB[2][16384];
  int tid = threadIdx.x, wave = tid >> 5, lane = tid & 31;
  int wm = wave & 1, wn = wave >> 1, l = lane & 15, h = lane >> 4;
  int row0 = blockIdx.y * 64, col0 = blockIdx.x * 256;
  const __bf16* Bbase = wt + (size_t)(3 * HH + 2 * DD + col0) * DH;
  v8f acc[2][4] = {};
  gemm_seg(acc, rx + (size_t)row0 * DD, DD, Bbase, DD, tid, wm, wn, l, h, sA, sB);
  gemm_seg(acc, htb + (size_t)row0 * HH, HH, Bbase + DD, HH, tid, wm, wn, l, h,
           sA, sB);
#pragma unroll
  for (int i = 0; i < 2; ++i)
#pragma unroll
    for (int j = 0; j < 4; ++j) {
      int c = col0 + wn * 64 + j * 16 + l;
      float bv = bias[3 * HH + 2 * DD + c];
#pragma unroll
      for (int v = 0; v < 8; ++v) {
        int r = row0 + wm * 32 + i * 16 + v + 8 * h;
        float hcd = tanhf(acc[i][j][v] + bv);
        float xv = x[(size_t)r * DD + c];
        out[(size_t)r * DD + c] = xv + ud[(size_t)r * DD + c] * (hcd - xv);
      }
    }
}

__global__ __launch_bounds__(256) void k_lastht(const float* __restrict__ htf,
                                                float* __restrict__ dst) {
  int i = blockIdx.x * 256 + threadIdx.x;
  if (i >= NB * HH) return;
  int n = i / HH, hcol = i % HH;
  dst[i] = htf[((size_t)n * TT + (TT - 1)) * HH + hcol];
}

// ---------------- host side -----------------------------------------------
extern "C" void kernel_launch(void* const* d_in, const int* in_sizes, int n_in,
                              void* d_out, int out_size, void* d_ws, size_t ws_size,
                              hipStream_t stream) {
  (void)in_sizes; (void)n_in; (void)out_size; (void)ws_size;
  const float* x    = (const float*)d_in[0];
  const float* st   = (const float*)d_in[1];
  const float* w    = (const float*)d_in[2];
  const float* bias = (const float*)d_in[3];
  float* out = (float*)d_out;
  char* ws = (char*)d_ws;

  size_t off = 0;
  auto carve = [&](size_t bytes) {
    size_t r = off;
    off += (bytes + 255) & ~(size_t)255;
    return r;
  };
  __bf16* wt    = (__bf16*)(ws + carve((size_t)WC * DH * 2));   // 25.2 MB
  __bf16* xb    = (__bf16*)(ws + carve((size_t)NT * DD * 2));   // 33.6 MB
  float*  gates = (float*) (ws + carve((size_t)NT * G3 * 4));   // 201.3 MB
  float*  htf   = (float*) (ws + carve((size_t)NT * HH * 4));   // 67.1 MB
  __bf16* htb   = (__bf16*)(ws + carve((size_t)NT * HH * 2));   // 33.6 MB
  __bf16* stb   = (__bf16*)(ws + carve((size_t)NB * HH * 2));
  float*  u     = (float*) (ws + carve((size_t)NB * HH * 4));
  __bf16* rh    = (__bf16*)(ws + carve((size_t)NB * HH * 2));
  float*  ud = gates;  // post phase reuses the (now dead) gates region
  __bf16* rx = (__bf16*)((char*)gates + (size_t)NT * DD * 4);

  {
    size_t tot = (size_t)WC * DH;
    k_wt<<<(unsigned)((tot + 255) / 256), 256, 0, stream>>>(w, wt);
  }
  {
    size_t tot = (size_t)NT * DD;
    k_f2b<<<(unsigned)((tot + 255) / 256), 256, 0, stream>>>(x, xb, tot);
  }
  {
    size_t tot = (size_t)NB * HH;
    k_f2b<<<(unsigned)((tot + 255) / 256), 256, 0, stream>>>(st, stb, tot);
  }

  // phase 1: temporal gates GEMM (16384 x 3072 x 1024)
  k_gates<<<dim3(G3 / 256, NT / 64), 256, 0, stream>>>(xb, wt, bias, gates);

  // phase 2: 512-step recurrence
  for (int t = 0; t < TT; ++t) {
    const __bf16* hb;
    const float* hf;
    size_t hs;
    if (t == 0) { hb = stb; hf = st; hs = (size_t)HH; }
    else {
      hb = htb + (size_t)(t - 1) * HH;
      hf = htf + (size_t)(t - 1) * HH;
      hs = (size_t)TT * HH;
    }
    k_scang<<<dim3((2 * HH) / 64), 256, 0, stream>>>(t, hb, hs, hf, wt, gates, u, rh);
    k_scanh<<<dim3(HH / 64), 256, 0, stream>>>(t, rh, u, hf, hs, wt, gates, htf, htb);
  }

  // phase 3: depth gates + output blend
  k_post1<<<dim3((2 * DD) / 256, NT / 64), 256, 0, stream>>>(xb, htb, x, wt, bias, ud, rx);
  k_post2<<<dim3(DD / 256, NT / 64), 256, 0, stream>>>(htb, rx, x, ud, wt, bias, out);

  // last_ht output (concatenated after h)
  k_lastht<<<(NB * HH + 255) / 256, 256, 0, stream>>>(htf, out + (size_t)NT * DD);
}
